// GNNcluster_75368086110727
// MI455X (gfx1250) — compile-verified, matrix operand order checked by reading, not run on hardware
//
#include <hip/hip_runtime.h>
#include <hip/hip_bf16.h>

// ---------------------------------------------------------------------------
// DiffPool-style GNN for MI455X (gfx1250, wave32).
// Heavy GEMMs use v_wmma_f32_16x16x32_bf16 with f32 accumulate; edge
// scatter-mean uses coalesced float4 gathers + global_atomic_add_f32.
// ---------------------------------------------------------------------------

typedef float v8f  __attribute__((ext_vector_type(8)));
typedef __bf16 v16bf __attribute__((ext_vector_type(16)));

#define TM 64        // rows per block
#define TN 64        // cols per block
#define TK 32        // K per iteration (one wmma)
#define LSTR 36      // LDS row stride in bf16 elems (72B: 8B-aligned, low conflict)

static __device__ inline unsigned pack_bf16(float lo, float hi) {
    union { __bf16 b[2]; unsigned u; } p;
    p.b[0] = (__bf16)lo;
    p.b[1] = (__bf16)hi;
    return p.u;
}

// ---------------------------------------------------------------------------
// deg[dst] += 1 over edges
__global__ void deg_kernel(const int* __restrict__ dst, float* __restrict__ deg, int E) {
    int e = blockIdx.x * blockDim.x + threadIdx.x;
    if (e < E) atomicAdd(&deg[dst[e]], 1.0f);
}

// deg := 1 / max(deg, 1)
__global__ void recip_kernel(float* __restrict__ deg, int N) {
    int i = blockIdx.x * blockDim.x + threadIdx.x;
    if (i < N) deg[i] = 1.0f / fmaxf(deg[i], 1.0f);
}

// agg[dst] += h[src]; perEdge = H/4 threads per edge, float4 per thread
__global__ __launch_bounds__(256) void scatter_add_kernel(
    const float* __restrict__ h,
    const int* __restrict__ src, const int* __restrict__ dst,
    float* __restrict__ agg, int E, int H) {
    int perEdge = H >> 2;
    int edgesPerBlk = blockDim.x / perEdge;
    int e = blockIdx.x * edgesPerBlk + threadIdx.x / perEdge;
    if (e >= E) return;
    int c = (threadIdx.x % perEdge) * 4;
    int s = src[e], d = dst[e];
    const float4 v = *(const float4*)(h + (size_t)s * H + c);
    float* p = agg + (size_t)d * H + c;
    atomicAdd(p + 0, v.x);
    atomicAdd(p + 1, v.y);
    atomicAdd(p + 2, v.z);
    atomicAdd(p + 3, v.w);
}

// ---------------------------------------------------------------------------
// OUT = [relu]( (A1 * rdeg_row) @ W1  +  A2 @ W2  +  bias )
// A1,A2: [M,K] f32; W1,W2: [K,HO] f32; converted to bf16 in LDS, WMMA f32 acc.
__global__ __launch_bounds__(256) void sage_gemm_bf16_kernel(
    const float* __restrict__ A1, const float* __restrict__ rdeg,
    const float* __restrict__ A2,
    const float* __restrict__ W1, const float* __restrict__ W2,
    const float* __restrict__ bias, float* __restrict__ OUT,
    int M, int K, int HO, int do_relu)
{
    __shared__ __bf16 lA1[TM * LSTR];
    __shared__ __bf16 lA2[TM * LSTR];
    __shared__ __bf16 lW1[TN * LSTR];   // stored transposed: [col][k]
    __shared__ __bf16 lW2[TN * LSTR];

    const int tid  = threadIdx.x;
    const int wave = tid >> 5;
    const int lane = tid & 31;
    const int half = lane >> 4;
    const int l16  = lane & 15;

    const int rowBlock = blockIdx.x * TM;
    const int colBlock = blockIdx.y * TN;

    const int wr = wave & 3;            // 4 row-waves of 16 rows
    const int wc = wave >> 2;           // 2 col-waves of 32 cols
    const int waveRow = wr * 16;
    const int waveCol = wc * 32;

    v8f acc0 = {0.f,0.f,0.f,0.f,0.f,0.f,0.f,0.f};
    v8f acc1 = {0.f,0.f,0.f,0.f,0.f,0.f,0.f,0.f};

    // A loader: 4 threads/row x 8 k each (two float4 loads per matrix)
    const int ar = tid >> 2;
    const int ak = (tid & 3) * 8;
    // W loader: 8 threads/k-row x 8 cols each
    const int wk  = tid >> 3;
    const int wcl = (tid & 7) * 8;

    const bool rowOk = (rowBlock + ar) < M;
    const float scale = rowOk ? rdeg[rowBlock + ar] : 0.f;
    const bool colOk = (colBlock + wcl + 7) < HO;

    for (int kk = 0; kk < K; kk += TK) {
        // ---- stage A tiles: vectorized b128 loads, packed bf16 b64 stores ----
        {
            int gr = rowBlock + ar;
            float4 a1lo = {0,0,0,0}, a1hi = {0,0,0,0};
            float4 a2lo = {0,0,0,0}, a2hi = {0,0,0,0};
            if (rowOk) {
                const float4* pA1 = (const float4*)(A1 + (size_t)gr * K + kk + ak);
                const float4* pA2 = (const float4*)(A2 + (size_t)gr * K + kk + ak);
                a1lo = pA1[0]; a1hi = pA1[1];
                a2lo = pA2[0]; a2hi = pA2[1];
            }
            uint2* d1 = (uint2*)&lA1[ar * LSTR + ak];
            uint2* d2 = (uint2*)&lA2[ar * LSTR + ak];
            d1[0] = make_uint2(pack_bf16(a1lo.x * scale, a1lo.y * scale),
                               pack_bf16(a1lo.z * scale, a1lo.w * scale));
            d1[1] = make_uint2(pack_bf16(a1hi.x * scale, a1hi.y * scale),
                               pack_bf16(a1hi.z * scale, a1hi.w * scale));
            d2[0] = make_uint2(pack_bf16(a2lo.x, a2lo.y), pack_bf16(a2lo.z, a2lo.w));
            d2[1] = make_uint2(pack_bf16(a2hi.x, a2hi.y), pack_bf16(a2hi.z, a2hi.w));
        }
        // ---- stage W tiles: b128 loads, transposed b16 scatter into LDS ----
        {
            float4 w1lo = {0,0,0,0}, w1hi = {0,0,0,0};
            float4 w2lo = {0,0,0,0}, w2hi = {0,0,0,0};
            if (colOk) {
                const float4* pW1 = (const float4*)(W1 + (size_t)(kk + wk) * HO + colBlock + wcl);
                const float4* pW2 = (const float4*)(W2 + (size_t)(kk + wk) * HO + colBlock + wcl);
                w1lo = pW1[0]; w1hi = pW1[1];
                w2lo = pW2[0]; w2hi = pW2[1];
            }
            float w1v[8] = {w1lo.x, w1lo.y, w1lo.z, w1lo.w, w1hi.x, w1hi.y, w1hi.z, w1hi.w};
            float w2v[8] = {w2lo.x, w2lo.y, w2lo.z, w2lo.w, w2hi.x, w2hi.y, w2hi.z, w2hi.w};
            #pragma unroll
            for (int i = 0; i < 8; ++i) {
                lW1[(wcl + i) * LSTR + wk] = (__bf16)w1v[i];
                lW2[(wcl + i) * LSTR + wk] = (__bf16)w2v[i];
            }
        }
        __syncthreads();

        // ---- fragments per ISA 16-bit layout:
        // lanes 0-15: K 0..7 & 16..23 ; lanes 16-31: K 8..15 & 24..31
        const int kb = half * 8;
        v16bf a1, a2, b00, b01, b10, b11;
        const __bf16* pa1 = &lA1[(waveRow + l16) * LSTR];
        const __bf16* pa2 = &lA2[(waveRow + l16) * LSTR];
        const __bf16* p00 = &lW1[(waveCol + l16) * LSTR];
        const __bf16* p01 = &lW2[(waveCol + l16) * LSTR];
        const __bf16* p10 = &lW1[(waveCol + 16 + l16) * LSTR];
        const __bf16* p11 = &lW2[(waveCol + 16 + l16) * LSTR];
        #pragma unroll
        for (int i = 0; i < 8; ++i) {
            a1[i]  = pa1[kb + i];      a1[8 + i]  = pa1[kb + 16 + i];
            a2[i]  = pa2[kb + i];      a2[8 + i]  = pa2[kb + 16 + i];
            b00[i] = p00[kb + i];      b00[8 + i] = p00[kb + 16 + i];
            b01[i] = p01[kb + i];      b01[8 + i] = p01[kb + 16 + i];
            b10[i] = p10[kb + i];      b10[8 + i] = p10[kb + 16 + i];
            b11[i] = p11[kb + i];      b11[8 + i] = p11[kb + 16 + i];
        }
        acc0 = __builtin_amdgcn_wmma_f32_16x16x32_bf16(false, a1, false, b00, (short)0, acc0, false, false);
        acc0 = __builtin_amdgcn_wmma_f32_16x16x32_bf16(false, a2, false, b01, (short)0, acc0, false, false);
        acc1 = __builtin_amdgcn_wmma_f32_16x16x32_bf16(false, a1, false, b10, (short)0, acc1, false, false);
        acc1 = __builtin_amdgcn_wmma_f32_16x16x32_bf16(false, a2, false, b11, (short)0, acc1, false, false);
        __syncthreads();
    }

    // ---- epilogue: C/D layout — vgpr v: M = v + 8*half, N = lane&15 ----
    #pragma unroll
    for (int v = 0; v < 8; ++v) {
        int row = rowBlock + waveRow + v + 8 * half;
        if (row >= M) continue;
        int c0 = colBlock + waveCol + l16;
        int c1 = c0 + 16;
        if (c0 < HO) {
            float o = acc0[v] + bias[c0];
            OUT[(size_t)row * HO + c0] = do_relu ? fmaxf(o, 0.f) : o;
        }
        if (c1 < HO) {
            float o = acc1[v] + bias[c1];
            OUT[(size_t)row * HO + c1] = do_relu ? fmaxf(o, 0.f) : o;
        }
    }
}

// ---------------------------------------------------------------------------
// s2 = softmax( (agg*rdeg) @ Wl + b + hprev @ Wr ), K=256 -> C=8.
// One wave per node; cross-lane reduce via shfl_xor.
__global__ __launch_bounds__(256) void assign2_softmax_kernel(
    const float* __restrict__ agg, const float* __restrict__ rdeg,
    const float* __restrict__ hprev,
    const float* __restrict__ Wl, const float* __restrict__ Wr,
    const float* __restrict__ b, float* __restrict__ sOut, int N, int K)
{
    __shared__ float lWl[256 * 8];
    __shared__ float lWr[256 * 8];
    for (int i = threadIdx.x; i < K * 8; i += blockDim.x) {
        lWl[i] = Wl[i];
        lWr[i] = Wr[i];
    }
    __syncthreads();

    int wave = threadIdx.x >> 5;
    int lane = threadIdx.x & 31;
    int n = blockIdx.x * 8 + wave;
    if (n >= N) return;

    float acc[8] = {0.f,0.f,0.f,0.f,0.f,0.f,0.f,0.f};
    float rd = rdeg[n];
    for (int k = lane; k < K; k += 32) {
        float a = agg[(size_t)n * K + k] * rd;
        float h = hprev[(size_t)n * K + k];
        #pragma unroll
        for (int c = 0; c < 8; ++c)
            acc[c] += a * lWl[k * 8 + c] + h * lWr[k * 8 + c];
    }
    #pragma unroll
    for (int c = 0; c < 8; ++c)
        #pragma unroll
        for (int off = 16; off > 0; off >>= 1)
            acc[c] += __shfl_xor(acc[c], off, 32);

    if (lane < 8) {
        float logits[8];
        float m = -1e30f;
        #pragma unroll
        for (int c = 0; c < 8; ++c) { logits[c] = acc[c] + b[c]; m = fmaxf(m, logits[c]); }
        float sum = 0.f;
        #pragma unroll
        for (int c = 0; c < 8; ++c) { logits[c] = __expf(logits[c] - m); sum += logits[c]; }
        sOut[(size_t)n * 8 + lane] = logits[lane] / sum;
    }
}

// ---------------------------------------------------------------------------
// x_pool[g,c,h] = sum_{n: batch[n]==g} s[n,c] * z[n,h]
// batch is sorted: binary-search per-graph node range; 8 node-chunks per graph.
__global__ __launch_bounds__(256) void pool_kernel(
    const float* __restrict__ s, const float* __restrict__ z,
    const int* __restrict__ batch, float* __restrict__ xpool, int N, int H)
{
    int g = blockIdx.x;
    int chunk = blockIdx.y, nch = gridDim.y;
    int lo = 0, hi = N;
    while (lo < hi) { int mid = (lo + hi) >> 1; if (batch[mid] < g) lo = mid + 1; else hi = mid; }
    int start = lo;
    hi = N;
    while (lo < hi) { int mid = (lo + hi) >> 1; if (batch[mid] < g + 1) lo = mid + 1; else hi = mid; }
    int end = lo;

    int t = threadIdx.x;           // blockDim.x == H
    float acc[8] = {0.f,0.f,0.f,0.f,0.f,0.f,0.f,0.f};
    for (int n = start + chunk; n < end; n += nch) {
        float zv = z[(size_t)n * H + t];
        const float* sp = s + (size_t)n * 8;
        #pragma unroll
        for (int c = 0; c < 8; ++c) acc[c] += sp[c] * zv;
    }
    #pragma unroll
    for (int c = 0; c < 8; ++c)
        atomicAdd(&xpool[(size_t)g * 8 * H + c * H + t], acc[c]);
}

// ---------------------------------------------------------------------------
// out[g] = (relu(x_pool[g] @ Wc1 + bc1) @ Wc2 + bc2)
__global__ __launch_bounds__(256) void classifier_kernel(
    const float* __restrict__ xpool, const float* __restrict__ Wc1,
    const float* __restrict__ bc1, const float* __restrict__ Wc2,
    const float* __restrict__ bc2, float* __restrict__ out, int CH, int H)
{
    __shared__ float xp[2048];
    __shared__ float red[256];
    int g = blockIdx.x, t = threadIdx.x;
    for (int i = t; i < CH; i += blockDim.x) xp[i] = xpool[(size_t)g * CH + i];
    __syncthreads();
    float acc = bc1[t];
    for (int k = 0; k < CH; ++k) acc += xp[k] * Wc1[(size_t)k * H + t];
    red[t] = fmaxf(acc, 0.f) * Wc2[t];
    __syncthreads();
    for (int s = 128; s > 0; s >>= 1) {
        if (t < s) red[t] += red[t + s];
        __syncthreads();
    }
    if (t == 0) out[g] = red[0] + bc2[0];
}

// ---------------------------------------------------------------------------
extern "C" void kernel_launch(void* const* d_in, const int* in_sizes, int n_in,
                              void* d_out, int out_size, void* d_ws, size_t ws_size,
                              hipStream_t stream) {
    const float* x     = (const float*)d_in[0];
    const int*   ei    = (const int*)d_in[1];
    const int*   batch = (const int*)d_in[2];
    const float* We1_l = (const float*)d_in[3];
    const float* be1   = (const float*)d_in[4];
    const float* We1_r = (const float*)d_in[5];
    const float* We2_l = (const float*)d_in[6];
    const float* be2   = (const float*)d_in[7];
    const float* We2_r = (const float*)d_in[8];
    const float* Wa1_l = (const float*)d_in[9];
    const float* ba1   = (const float*)d_in[10];
    const float* Wa1_r = (const float*)d_in[11];
    const float* Wa2_l = (const float*)d_in[12];
    const float* ba2   = (const float*)d_in[13];
    const float* Wa2_r = (const float*)d_in[14];
    const float* Wc1   = (const float*)d_in[15];
    const float* bc1   = (const float*)d_in[16];
    const float* Wc2   = (const float*)d_in[17];
    const float* bc2   = (const float*)d_in[18];
    float* out = (float*)d_out;

    const int N  = in_sizes[2];
    const int E  = in_sizes[1] / 2;
    const int IN = in_sizes[0] / N;
    const int H  = in_sizes[4];
    const int C  = in_sizes[13];
    const int G  = out_size;
    const int* src = ei;
    const int* dst = ei + E;

    // workspace carve-out
    float* ws = (float*)d_ws;
    size_t off = 0;
    auto carve = [&](size_t n) { float* p = ws + off; off += (n + 63) & ~(size_t)63; return p; };
    float* deg   = carve((size_t)N);
    float* agg   = carve((size_t)N * H);
    float* z1    = carve((size_t)N * H);
    float* s1    = carve((size_t)N * H);
    float* z2    = carve((size_t)N * H);
    float* s2    = carve((size_t)N * C);
    float* xpool = carve((size_t)G * C * H);

    const int T = 256;
    dim3 gemmGrid((N + TM - 1) / TM, H / TN);
    int edgesPerBlk = T / (H / 4);                 // 4 for H=256
    int scatBlocks = (E + edgesPerBlk - 1) / edgesPerBlk;

    // 1) degree + reciprocal
    hipMemsetAsync(deg, 0, (size_t)N * sizeof(float), stream);
    deg_kernel<<<(E + T - 1) / T, T, 0, stream>>>(dst, deg, E);
    recip_kernel<<<(N + T - 1) / T, T, 0, stream>>>(deg, N);

    // 2) scatter-mean of x (shared by embed1 + assign1)
    hipMemsetAsync(agg, 0, (size_t)N * H * sizeof(float), stream);
    scatter_add_kernel<<<scatBlocks, T, 0, stream>>>(x, src, dst, agg, E, IN);

    // 3) embed1: z1 = relu(agg/deg @ We1_l + be1 + x @ We1_r)
    sage_gemm_bf16_kernel<<<gemmGrid, T, 0, stream>>>(agg, deg, x, We1_l, We1_r, be1, z1, N, IN, H, 1);
    // 4) assign1: s1 = agg/deg @ Wa1_l + ba1 + x @ Wa1_r
    sage_gemm_bf16_kernel<<<gemmGrid, T, 0, stream>>>(agg, deg, x, Wa1_l, Wa1_r, ba1, s1, N, IN, H, 0);

    // 5) embed2: z2 = relu(scatter(z1)/deg @ We2_l + be2 + z1 @ We2_r)
    hipMemsetAsync(agg, 0, (size_t)N * H * sizeof(float), stream);
    scatter_add_kernel<<<scatBlocks, T, 0, stream>>>(z1, src, dst, agg, E, H);
    sage_gemm_bf16_kernel<<<gemmGrid, T, 0, stream>>>(agg, deg, z1, We2_l, We2_r, be2, z2, N, H, H, 1);

    // 6) assign2 + softmax: s2 = softmax(scatter(s1)/deg @ Wa2_l + ba2 + s1 @ Wa2_r)
    hipMemsetAsync(agg, 0, (size_t)N * H * sizeof(float), stream);
    scatter_add_kernel<<<scatBlocks, T, 0, stream>>>(s1, src, dst, agg, E, H);
    assign2_softmax_kernel<<<(N + 7) / 8, T, 0, stream>>>(agg, deg, s1, Wa2_l, Wa2_r, ba2, s2, N, H);

    // 7) diffpool outer-product reduction
    hipMemsetAsync(xpool, 0, (size_t)G * C * H * sizeof(float), stream);
    pool_kernel<<<dim3(G, 8), H, 0, stream>>>(s2, z2, batch, xpool, N, H);

    // 8) classifier head
    classifier_kernel<<<G, H, 0, stream>>>(xpool, Wc1, bc1, Wc2, bc2, out, C * H, H);
}
